// Model_55963423867304
// MI455X (gfx1250) — matrix operand order repaired and checked by reading, not compiled, not run on hardware
//
#include <hip/hip_runtime.h>
#include <stdint.h>

// ---------------------------------------------------------------------------
// Problem dims (fixed by reference)
// ---------------------------------------------------------------------------
#define B_DIM   256
#define J_DIM   128
#define I_DIM   64
#define E_DIM   512
#define H_DIM   512
#define M_DIM   (B_DIM * J_DIM)   // 32768 rows for both GEMMs
#define N2_DIM  256               // jump columns actually used (idx in [1,255])
#define KC1     (E_DIM / 32)      // 16 k-chunks of 32 for GEMM1
#define KC2     (H_DIM / 32)      // 16 k-chunks of 32 for GEMM2
#define EPS_F   1e-29f

// ---------------------------------------------------------------------------
// WMMA types (CDNA5 wave32: 16x16x32 bf16 -> f32, A/B = v16bf, C/D = v8f)
// ---------------------------------------------------------------------------
typedef __attribute__((ext_vector_type(16))) __bf16 v16bf;
typedef __attribute__((ext_vector_type(8)))  float  v8f;

union V16BF {
    unsigned short u[16];
    uint32_t       d[8];
    uint4          q[2];
    v16bf          v;
};

__device__ __forceinline__ v8f wmma_bf16(const v16bf& a, const v16bf& b, v8f c) {
    // (neg_a, A, neg_b, B, c_mod, C, reuse_a, reuse_b)
    return __builtin_amdgcn_wmma_f32_16x16x32_bf16(false, a, false, b,
                                                   (short)0, c, false, false);
}

// Split f32 into bf16 hi (truncate) + bf16 lo (residual): x ~= hi + lo
__device__ __forceinline__ void bf16_hilo(float x, unsigned short& h, unsigned short& l) {
    uint32_t u = __float_as_uint(x);
    h = (unsigned short)(u >> 16);
    float hf = __uint_as_float(u & 0xFFFF0000u);
    l = (unsigned short)(__float_as_uint(x - hf) >> 16);
}

// Load a 16x32 f32 A-fragment (row-major, leading dim ld) and convert to
// bf16 hi/lo fragments in the CDNA5 wave32 A layout:
//   lane<16 : row=lane,    K in [kb..kb+7] and [16+kb..16+kb+7], kb=0
//   lane>=16: row=lane-16, kb=8
__device__ __forceinline__ void load_a_frag(const float* __restrict__ A, size_t ld,
                                            int m0, int kk, int lane,
                                            V16BF& ah, V16BF& al) {
    int r  = lane & 15;
    int kb = (lane < 16) ? 0 : 8;
    const float* p = A + (size_t)(m0 + r) * ld + (size_t)(kk + kb);
    float4 a0 = *(const float4*)(p);
    float4 a1 = *(const float4*)(p + 4);
    float4 a2 = *(const float4*)(p + 16);
    float4 a3 = *(const float4*)(p + 20);
    float av[16] = {a0.x, a0.y, a0.z, a0.w, a1.x, a1.y, a1.z, a1.w,
                    a2.x, a2.y, a2.z, a2.w, a3.x, a3.y, a3.z, a3.w};
#pragma unroll
    for (int j = 0; j < 16; ++j) bf16_hilo(av[j], ah.u[j], al.u[j]);
}

// Load a prepacked B fragment (8 dwords per lane, contiguous -> 2x b128)
__device__ __forceinline__ void load_b_frag(const uint32_t* __restrict__ W,
                                            int ntile, int kc, int nKC, int lane,
                                            V16BF& b) {
    const uint32_t* p = W + ((((size_t)ntile * nKC) + kc) * 32 + (size_t)lane) * 8;
    b.q[0] = *(const uint4*)(p);
    b.q[1] = *(const uint4*)(p + 4);
}

// ---------------------------------------------------------------------------
// Prep: convert row-major f32 weights [K x ldN] into fragment-native packed
// bf16 hi/lo buffers. One thread per (tile, kchunk, lane, vgpr).
// B layout (32x16, 16-bit): lane<16 -> n=lane, K=2v,2v+1 (v<4), 16+2(v-4).. (v>=4) ; lane>=16 -> K+=8.
// ---------------------------------------------------------------------------
__global__ void prep_w(const float* __restrict__ W, int ldN, int nKC, int total,
                       uint32_t* __restrict__ WH, uint32_t* __restrict__ WL) {
    int tid = blockIdx.x * blockDim.x + threadIdx.x;
    if (tid >= total) return;
    int v    = tid & 7;
    int lane = (tid >> 3) & 31;
    int rest = tid >> 8;
    int kc   = rest % nKC;
    int tile = rest / nKC;
    int n  = tile * 16 + (lane & 15);
    int kb = (lane < 16) ? 0 : 8;
    int k  = kc * 32 + kb + 2 * v + ((v >= 4) ? 8 : 0);
    float f0 = W[(size_t)k * ldN + n];
    float f1 = W[(size_t)(k + 1) * ldN + n];
    unsigned short h0, l0, h1, l1;
    bf16_hilo(f0, h0, l0);
    bf16_hilo(f1, h1, l1);
    size_t out = (((size_t)tile * nKC + kc) * 32 + lane) * 8 + v;
    WH[out] = ((uint32_t)h1 << 16) | (uint32_t)h0;
    WL[out] = ((uint32_t)l1 << 16) | (uint32_t)l0;
}

// ---------------------------------------------------------------------------
// p0 = sigmoid(tanh(y_null @ W_trans + b_trans) @ W_p0 + b_p0)  (scalar)
// ---------------------------------------------------------------------------
__global__ __launch_bounds__(256) void p0_kernel(
    const float* __restrict__ ynull, const float* __restrict__ W1,
    const float* __restrict__ b1, const float* __restrict__ Wp0,
    const float* __restrict__ bp0, float* __restrict__ p0out) {
    __shared__ float s[256];
    float part = 0.f;
    for (int h = threadIdx.x; h < H_DIM; h += 256) {
        float acc = b1[h];
        for (int e = 0; e < E_DIM; ++e) acc += ynull[e] * W1[(size_t)e * H_DIM + h];
        part += tanhf(acc) * Wp0[h];
    }
    s[threadIdx.x] = part;
    __syncthreads();
    for (int st = 128; st > 0; st >>= 1) {
        if (threadIdx.x < st) s[threadIdx.x] += s[threadIdx.x + st];
        __syncthreads();
    }
    if (threadIdx.x == 0) {
        float z = s[0] + bp0[0];
        p0out[0] = 1.0f / (1.0f + expf(-z));   // ALPHA_P0 == 1.0
    }
}

// ---------------------------------------------------------------------------
// GEMM1: ts = tanh(A[32768x512] @ W1[512x512] + b1)
// grid = 2048 blocks (16 rows each), 8 waves/block, wave -> 4 N-tiles (64 cols)
// Split-bf16: 3 WMMAs per tile per k-chunk.
// ---------------------------------------------------------------------------
__global__ __launch_bounds__(256) void gemm1_tanh(
    const float* __restrict__ A, const uint32_t* __restrict__ WH,
    const uint32_t* __restrict__ WL, const float* __restrict__ bias,
    float* __restrict__ ts) {
    int wave = threadIdx.x >> 5;
    int lane = threadIdx.x & 31;
    int m0   = blockIdx.x * 16;

    v8f acc[4];
#pragma unroll
    for (int t = 0; t < 4; ++t)
#pragma unroll
        for (int v = 0; v < 8; ++v) acc[t][v] = 0.f;

    for (int kc = 0; kc < KC1; ++kc) {
        V16BF ah, al;
        load_a_frag(A, E_DIM, m0, kc * 32, lane, ah, al);
#pragma unroll
        for (int t = 0; t < 4; ++t) {
            int ntile = wave * 4 + t;
            V16BF bh, bl;
            load_b_frag(WH, ntile, kc, KC1, lane, bh);
            load_b_frag(WL, ntile, kc, KC1, lane, bl);
            acc[t] = wmma_bf16(ah.v, bh.v, acc[t]);
            acc[t] = wmma_bf16(ah.v, bl.v, acc[t]);
            acc[t] = wmma_bf16(al.v, bh.v, acc[t]);
        }
    }

    int rbase = m0 + ((lane >> 4) << 3);   // C layout: lanes>=16 hold M+8
#pragma unroll
    for (int t = 0; t < 4; ++t) {
        int col  = (wave * 4 + t) * 16 + (lane & 15);
        float bv = bias[col];
#pragma unroll
        for (int v = 0; v < 8; ++v)
            ts[(size_t)(rbase + v) * H_DIM + col] = tanhf(acc[t][v] + bv);
    }
}

// ---------------------------------------------------------------------------
// GEMM2 + fused windowed softmax:
//   logits = ts @ W_jump + b_jump  (cols 0..255; col 256 provably unused)
//   jw[b,i,k] = softmax_{k}(logits[b,i, 128-i+k]) * (1 - p0)
// grid = 2048 blocks (16 rows), 8 waves x 2 N-tiles = 256 cols into LDS.
// ---------------------------------------------------------------------------
__global__ __launch_bounds__(256) void gemm2_jw(
    const float* __restrict__ ts, const uint32_t* __restrict__ WH,
    const uint32_t* __restrict__ WL, const float* __restrict__ bias,
    const float* __restrict__ p0buf, float* __restrict__ jw) {
    __shared__ float s_log[16][N2_DIM];   // 16 KB
    __shared__ float s_scale[16];

    int wave = threadIdx.x >> 5;
    int lane = threadIdx.x & 31;
    int m0   = blockIdx.x * 16;

    v8f acc[2];
#pragma unroll
    for (int t = 0; t < 2; ++t)
#pragma unroll
        for (int v = 0; v < 8; ++v) acc[t][v] = 0.f;

    for (int kc = 0; kc < KC2; ++kc) {
        V16BF ah, al;
        load_a_frag(ts, H_DIM, m0, kc * 32, lane, ah, al);
#pragma unroll
        for (int t = 0; t < 2; ++t) {
            int ntile = wave * 2 + t;
            V16BF bh, bl;
            load_b_frag(WH, ntile, kc, KC2, lane, bh);
            load_b_frag(WL, ntile, kc, KC2, lane, bl);
            acc[t] = wmma_bf16(ah.v, bh.v, acc[t]);
            acc[t] = wmma_bf16(ah.v, bl.v, acc[t]);
            acc[t] = wmma_bf16(al.v, bh.v, acc[t]);
        }
    }

    int rbase = (lane >> 4) << 3;
#pragma unroll
    for (int t = 0; t < 2; ++t) {
        int col  = (wave * 2 + t) * 16 + (lane & 15);
        float bv = bias[col];
#pragma unroll
        for (int v = 0; v < 8; ++v) s_log[rbase + v][col] = acc[t][v] + bv;
    }
    __syncthreads();

    int bidx  = m0 >> 7;     // batch index (16-row blocks never straddle J=128)
    int ibase = m0 & 127;

    // Phase 1: one leader thread per row: windowed max + exp + sum
    if (threadIdx.x < 16) {
        int r  = threadIdx.x;
        int i  = ibase + r;
        int w0 = 128 - i;                 // window start in [1,128]
        float mx = -3.4e38f;
        for (int k = 0; k < 128; ++k) mx = fmaxf(mx, s_log[r][w0 + k]);
        float sum = 0.f;
        for (int k = 0; k < 128; ++k) {
            float e = expf(s_log[r][w0 + k] - mx);
            s_log[r][w0 + k] = e;
            sum += e;
        }
        s_scale[r] = (1.0f - p0buf[0]) / sum;
    }
    __syncthreads();

    // Phase 2: coalesced store of jw[b, i, 0..127]
#pragma unroll
    for (int it = 0; it < 8; ++it) {
        int e = threadIdx.x + 256 * it;   // 2048 = 16 rows * 128 cols
        int r = e >> 7;
        int k = e & 127;
        int i = ibase + r;
        jw[((size_t)bidx * J_DIM + i) * J_DIM + k] = s_log[r][(128 - i) + k] * s_scale[r];
    }
}

// ---------------------------------------------------------------------------
// HMM forward recursion. One block per batch, 128 threads.
// f@T collapses to: g = f_lo + f_hi ; out_lo = g @ jw ; out_hi = g * p0.
// jw column k lives in this thread's registers (fully unrolled 128-FMA dot).
// Records log(sum(masked f)) at t = source_len-1, then exits.
// ---------------------------------------------------------------------------
__global__ __launch_bounds__(128) void hmm_forward(
    const float* __restrict__ jw, const float* __restrict__ emission,
    const float* __restrict__ p0buf, const int* __restrict__ slen,
    const int* __restrict__ tlen, float* __restrict__ out) {
    __shared__ float s_g[128];
    __shared__ float s_red[128];

    int b = blockIdx.x;
    int k = threadIdx.x;
    float p0 = p0buf[0];

    const float* jwb = jw + (size_t)b * J_DIM * J_DIM;
    float col[128];
#pragma unroll
    for (int i = 0; i < 128; ++i) col[i] = jwb[(size_t)i * J_DIM + k];   // coalesced

    const float* em = emission + (size_t)b * (2 * J_DIM) * I_DIM;       // [2J][I]
    int sl = slen[b];
    int tl = tlen[b];

    float fl = em[(size_t)k * I_DIM];            // f0 = emission[:, :, 0]
    float fh = em[(size_t)(k + 128) * I_DIM];

    for (int t = 0; t < I_DIM; ++t) {
        if (t > 0) {
            float gk = fl + fh;
            s_g[k] = gk;
            __syncthreads();
            float a = 0.f;
#pragma unroll
            for (int i = 0; i < 128; ++i) a += s_g[i] * col[i];
            fl = a * em[(size_t)k * I_DIM + t];
            fh = gk * p0 * em[(size_t)(k + 128) * I_DIM + t];
            __syncthreads();
        }
        if (t == sl - 1) {
            s_red[k] = (k < tl) ? (fl + fh) : 0.f;
            __syncthreads();
            for (int st = 64; st > 0; st >>= 1) {
                if (k < st) s_red[k] += s_red[k + st];
                __syncthreads();
            }
            if (k == 0) out[b] = logf(s_red[0] + EPS_F);
            break;
        }
    }
}

// ---------------------------------------------------------------------------
// Launch
// ---------------------------------------------------------------------------
extern "C" void kernel_launch(void* const* d_in, const int* in_sizes, int n_in,
                              void* d_out, int out_size, void* d_ws, size_t ws_size,
                              hipStream_t stream) {
    const float* y_hidden = (const float*)d_in[0];
    const float* y_null   = (const float*)d_in[1];
    const float* W_trans  = (const float*)d_in[2];
    const float* b_trans  = (const float*)d_in[3];
    const float* W_jump   = (const float*)d_in[4];
    const float* b_jump   = (const float*)d_in[5];
    const float* W_p0     = (const float*)d_in[6];
    const float* b_p0     = (const float*)d_in[7];
    const float* emission = (const float*)d_in[8];
    const int*   slen     = (const int*)d_in[9];
    const int*   tlen     = (const int*)d_in[10];
    float* out = (float*)d_out;

    // Workspace carve-up (~86 MB total)
    char*  ws  = (char*)d_ws;
    size_t off = 0;
    auto carve = [&](size_t bytes) -> void* {
        void* p = (void*)(ws + off);
        off = (off + bytes + 255) & ~(size_t)255;
        return p;
    };
    const int totalW1 = 32 * KC1 * 32 * 8;   // 131072 dwords
    const int totalW2 = 16 * KC2 * 32 * 8;   //  65536 dwords
    uint32_t* WH1 = (uint32_t*)carve((size_t)totalW1 * 4);
    uint32_t* WL1 = (uint32_t*)carve((size_t)totalW1 * 4);
    uint32_t* WH2 = (uint32_t*)carve((size_t)totalW2 * 4);
    uint32_t* WL2 = (uint32_t*)carve((size_t)totalW2 * 4);
    float*    p0b = (float*)carve(256);
    float*    ts  = (float*)carve((size_t)M_DIM * H_DIM * 4);          // 64 MB
    float*    jw  = (float*)carve((size_t)B_DIM * J_DIM * J_DIM * 4);  // 16 MB

    prep_w<<<(totalW1 + 255) / 256, 256, 0, stream>>>(W_trans, H_DIM, KC1, totalW1, WH1, WL1);
    prep_w<<<(totalW2 + 255) / 256, 256, 0, stream>>>(W_jump, 257, KC2, totalW2, WH2, WL2);
    p0_kernel<<<1, 256, 0, stream>>>(y_null, W_trans, b_trans, W_p0, b_p0, p0b);
    gemm1_tanh<<<M_DIM / 16, 256, 0, stream>>>(y_hidden, WH1, WL1, b_trans, ts);
    gemm2_jw<<<M_DIM / 16, 256, 0, stream>>>(ts, WH2, WL2, b_jump, p0b, jw);
    hmm_forward<<<B_DIM, 128, 0, stream>>>(jw, emission, p0b, slen, tlen, out);
}
